// GAT_87617332838818
// MI455X (gfx1250) — compile-verified, matrix operand order checked by reading, not run on hardware
//
#include <hip/hip_runtime.h>

// ---------------------------------------------------------------------------
// Types for CDNA5 WMMA (wave32): v_wmma_f32_16x16x32_bf16
// ---------------------------------------------------------------------------
typedef __attribute__((ext_vector_type(16))) __bf16 v16bf;
typedef __attribute__((ext_vector_type(8)))  __bf16 v8bf;
typedef __attribute__((ext_vector_type(8)))  float  v8f;
typedef int v4i_vs __attribute__((vector_size(16)));   // matches async-LDS builtin param

// Ordered-int encoding of float for atomicMax-based segment max.
// Monotonic: a < b (float)  <=>  ord(a) < ord(b) (int). Self-inverse.
__device__ __forceinline__ int f32_ord(float f) {
  int b = __float_as_int(f);
  return b < 0 ? (b ^ 0x7fffffff) : b;
}
__device__ __forceinline__ float ord_f32(int b) {
  b = b < 0 ? (b ^ 0x7fffffff) : b;
  return __int_as_float(b);
}
#define ORD_NEG_INF (-2139095041)   // f32_ord(-inf) = 0x807FFFFF

// f32 -> bf16 round-to-nearest-even (raw bits)
__device__ __forceinline__ unsigned short f2bf(float f) {
  unsigned u = __float_as_uint(f);
  u += 0x7fffu + ((u >> 16) & 1u);
  return (unsigned short)(u >> 16);
}

// ---------------------------------------------------------------------------
// Elementwise converts
// ---------------------------------------------------------------------------
__global__ void k_cvt_f32_bf16(const float* __restrict__ s,
                               unsigned short* __restrict__ d, int n) {
  int t = blockIdx.x * blockDim.x + threadIdx.x;
  if (t < n) d[t] = f2bf(s[t]);
}

// W flat [C][K] f32 -> bf16 [Cpad][K], zero rows c >= C
__global__ void k_cvt_w_pad(const float* __restrict__ W,
                            unsigned short* __restrict__ d,
                            int C, int Cpad, int K) {
  int t = blockIdx.x * blockDim.x + threadIdx.x;
  if (t >= Cpad * K) return;
  int c = t / K;
  d[t] = f2bf(c < C ? W[t] : 0.0f);
}

__global__ void k_fill_i32(int* __restrict__ p, int v, int n) {
  int t = blockIdx.x * blockDim.x + threadIdx.x;
  if (t < n) p[t] = v;
}

// ---------------------------------------------------------------------------
// GEMM: C[N x CT*16] = A[N x K] * B^T, B stored [CT*16][K] (W's natural flat
// [head][o][i] layout). Block = 8 waves (256 thr); each wave owns a 16-row
// slab and computes ALL CT column tiles, holding its A fragments in registers
// (CTx cut in A traffic). B (<=24KB) is staged once per block into LDS with
// the CDNA5 async-to-LDS path (ASYNCcnt) and consumed via ds_load_b128.
//
// bf16 fragment layout (ISA 7.12.2, 16-bit 16x32 A / 32x16 B):
//   lane l<16 : row/col l, K = k0 + {0..7, 16..23}
//   lane l+16 : row/col l, K = k0 + {8..15, 24..31}
// -> two contiguous 16-byte bf16 chunks per lane per k-step.
// ---------------------------------------------------------------------------
template <int K, int CT>
__global__ __launch_bounds__(256) void k_gemm_tile(
    const unsigned short* __restrict__ A, const unsigned short* __restrict__ B,
    float* __restrict__ C, int Nrows, int ldc, int colmax) {
  constexpr int KF = K / 32;            // WMMA k-steps
  constexpr int BE = CT * 16 * K;       // bf16 elements of B
  __shared__ unsigned short sB[BE];

  const int tid = threadIdx.x;

  // ---- cooperative stage of B into LDS (16B per transfer) ----
  constexpr int CHUNKS = BE / 8;        // 8 bf16 = 16 bytes
  for (int c = tid; c < CHUNKS; c += 256) {
#if __has_builtin(__builtin_amdgcn_global_load_async_to_lds_b128)
    __builtin_amdgcn_global_load_async_to_lds_b128(
        (v4i_vs*)(B + (size_t)c * 8),
        (v4i_vs*)(sB + (size_t)c * 8),
        0, 0);
#else
    *(v8bf*)(void*)(sB + (size_t)c * 8) =
        *(const v8bf*)(const void*)(B + (size_t)c * 8);
#endif
  }
#if __has_builtin(__builtin_amdgcn_s_wait_asynccnt)
  __builtin_amdgcn_s_wait_asynccnt(0);
#else
  asm volatile("s_wait_asynccnt 0" ::: "memory");
#endif
  __syncthreads();

  const int wave = tid >> 5;
  const int lane = tid & 31;
  const int half = lane >> 4;
  const int l    = lane & 15;
  const int row0 = (blockIdx.x * 8 + wave) * 16;
  if (row0 >= Nrows) return;            // wave-uniform: EXEC stays all-ones

  int ar = row0 + l;
  if (ar >= Nrows) ar = Nrows - 1;      // clamp (N%16==0 in practice)
  const unsigned short* arow = A + (size_t)ar * K + half * 8;

  v16bf afrag[KF];
#pragma unroll
  for (int kf = 0; kf < KF; ++kf) {
    v8bf lo = *(const v8bf*)(const void*)(arow + kf * 32);
    v8bf hi = *(const v8bf*)(const void*)(arow + kf * 32 + 16);
    afrag[kf] = __builtin_shufflevector(lo, hi,
        0,1,2,3,4,5,6,7,8,9,10,11,12,13,14,15);
  }

#pragma unroll
  for (int ct = 0; ct < CT; ++ct) {
    const unsigned short* bcol = sB + (size_t)(ct * 16 + l) * K + half * 8;
    v8f acc = {};
#pragma unroll
    for (int kf = 0; kf < KF; ++kf) {
      v8bf lo = *(const v8bf*)(const void*)(bcol + kf * 32);
      v8bf hi = *(const v8bf*)(const void*)(bcol + kf * 32 + 16);
      v16bf bv = __builtin_shufflevector(lo, hi,
          0,1,2,3,4,5,6,7,8,9,10,11,12,13,14,15);
      acc = __builtin_amdgcn_wmma_f32_16x16x32_bf16(
          /*neg_a=*/false, afrag[kf], /*neg_b=*/false, bv,
          /*c_mod=*/(short)0, acc, /*reuse_a=*/false, /*reuse_b=*/false);
    }
    // C/D layout: lane l<16 -> N=l, M=0..7 in acc[0..7]; lane>=16 -> M=8..15
    const int col = ct * 16 + l;
    if (col < colmax) {
      const int rb = row0 + half * 8;
      float* cp = C + (size_t)rb * ldc + col;
      if (rb + 8 <= Nrows) {            // fast path (always, when N%16==0)
#pragma unroll
        for (int r = 0; r < 8; ++r) cp[(size_t)r * ldc] = acc[r];
      } else {
#pragma unroll
        for (int r = 0; r < 8; ++r)
          if (rb + r < Nrows) cp[(size_t)r * ldc] = acc[r];
      }
    }
  }
}

// ---------------------------------------------------------------------------
// Attention: per-node dot pre-reduction  d1[h,n]=<hp[n,h*F:],A[h,:F]>, d2 same
// with A[h,F:2F].  Edge e score = leakyrelu(d1[h,dst]+d2[h,src], 0.01).
// ---------------------------------------------------------------------------
__global__ void k_node_dots(const float* __restrict__ hp,
                            const float* __restrict__ Aatt,
                            float* __restrict__ d1, float* __restrict__ d2,
                            int N, int F, int H, int ldh) {
  int t = blockIdx.x * blockDim.x + threadIdx.x;
  if (t >= N * H) return;
  int n = t / H, h = t - n * H;
  const float* row = hp + (size_t)n * ldh + h * F;
  const float* a1  = Aatt + (size_t)h * 2 * F;
  const float* a2  = a1 + F;
  float s1 = 0.f, s2 = 0.f;
  for (int o = 0; o < F; ++o) {
    float v = row[o];
    s1 = fmaf(v, a1[o], s1);
    s2 = fmaf(v, a2[o], s2);
  }
  d1[(size_t)h * N + n] = s1;
  d2[(size_t)h * N + n] = s2;
}

__device__ __forceinline__ float edge_score(const float* d1, const float* d2,
                                            int h, int N, int s, int d) {
  float v = d1[(size_t)h * N + d] + d2[(size_t)h * N + s];
  return v > 0.f ? v : 0.01f * v;   // leaky_relu(., 0.01)
}

__global__ void k_edge_max(const int* __restrict__ src, const int* __restrict__ dst,
                           const float* __restrict__ d1, const float* __restrict__ d2,
                           int* __restrict__ menc, int N, int E, int H) {
  int e = blockIdx.x * blockDim.x + threadIdx.x;
  if (e >= E) return;
  int s = src[e], d = dst[e];
  for (int h = 0; h < H; ++h)
    atomicMax(&menc[(size_t)h * N + d], f32_ord(edge_score(d1, d2, h, N, s, d)));
}

__global__ void k_edge_expsum(const int* __restrict__ src, const int* __restrict__ dst,
                              const float* __restrict__ d1, const float* __restrict__ d2,
                              const int* __restrict__ menc, float* __restrict__ ssum,
                              int N, int E, int H) {
  int e = blockIdx.x * blockDim.x + threadIdx.x;
  if (e >= E) return;
  int s = src[e], d = dst[e];
  for (int h = 0; h < H; ++h) {
    float m = ord_f32(menc[(size_t)h * N + d]);
    if (!__builtin_isfinite(m)) m = 0.0f;   // reference's where(isfinite)
    float p = __expf(edge_score(d1, d2, h, N, s, d) - m);
    atomicAdd(&ssum[(size_t)h * N + d], p);
  }
}

__global__ void k_edge_alpha(const int* __restrict__ src, const int* __restrict__ dst,
                             const float* __restrict__ d1, const float* __restrict__ d2,
                             const int* __restrict__ menc, const float* __restrict__ ssum,
                             float* __restrict__ asum, int N, int E, int H) {
  int e = blockIdx.x * blockDim.x + threadIdx.x;
  if (e >= E) return;
  int s = src[e], d = dst[e];
  for (int h = 0; h < H; ++h) {
    float m = ord_f32(menc[(size_t)h * N + d]);
    if (!__builtin_isfinite(m)) m = 0.0f;
    float p = __expf(edge_score(d1, d2, h, N, s, d) - m);
    atomicAdd(&asum[(size_t)h * N + d], p / ssum[(size_t)h * N + d]);
  }
}

// out[h,n,:] = elu(hp[n, h*F:] * asum[h,n])  -> bf16 next-layer input
__global__ void k_finalize_elu_bf16(const float* __restrict__ hp,
                                    const float* __restrict__ asum,
                                    unsigned short* __restrict__ out,
                                    int N, int F, int H) {
  int t = blockIdx.x * blockDim.x + threadIdx.x;
  int HF = F * H;
  if (t >= N * HF) return;
  int n = t / HF;
  int h = (t - n * HF) / F;
  float v = hp[t] * asum[(size_t)h * N + n];
  v = v > 0.f ? v : (__expf(v) - 1.0f);     // elu
  out[t] = f2bf(v);
}

// final logits (H=1, no activation)
__global__ void k_finalize_out(const float* __restrict__ hp,
                               const float* __restrict__ asum,
                               float* __restrict__ out, int N, int Ccls) {
  int t = blockIdx.x * blockDim.x + threadIdx.x;
  if (t >= N * Ccls) return;
  out[t] = hp[t] * asum[t / Ccls];
}

// ---------------------------------------------------------------------------
extern "C" void kernel_launch(void* const* d_in, const int* in_sizes, int n_in,
                              void* d_out, int out_size, void* d_ws, size_t ws_size,
                              hipStream_t stream) {
  const float* x  = (const float*)d_in[0];
  const float* W0 = (const float*)d_in[1];
  const float* A0 = (const float*)d_in[2];
  const float* W1 = (const float*)d_in[3];
  const float* A1 = (const float*)d_in[4];
  const float* W2 = (const float*)d_in[5];
  const float* A2 = (const float*)d_in[6];
  const int* src  = (const int*)d_in[7];
  const int* dst  = (const int*)d_in[8];

  const int IN = 128, HID = 32, H = 3, CLS = 40;
  const int N  = in_sizes[0] / IN;     // 50000 (multiple of 16)
  const int E  = in_sizes[7];          // 850000
  const int F0 = H * HID;              // 96
  const int CLSP = 48;                 // 40 padded to 16-col tiles

  // workspace carve-out (256B aligned)
  char* ws = (char*)d_ws;
  size_t off = 0;
  auto alloc = [&](size_t b) { size_t o = off; off += (b + 255) & ~(size_t)255; return o; };
  unsigned short* hb   = (unsigned short*)(ws + alloc((size_t)N * IN * 2)); // bf16 layer input
  float*          hp   = (float*)(ws + alloc((size_t)N * F0 * 4));          // f32 projections
  unsigned short* wb   = (unsigned short*)(ws + alloc((size_t)F0 * IN * 2));
  float*          dd1  = (float*)(ws + alloc((size_t)H * N * 4));
  float*          dd2  = (float*)(ws + alloc((size_t)H * N * 4));
  int*            menc = (int*)  (ws + alloc((size_t)H * N * 4));
  float*          ssum = (float*)(ws + alloc((size_t)H * N * 4));
  float*          asum = (float*)(ws + alloc((size_t)H * N * 4));

  const int TB = 256;
  auto cdiv = [](int a, int b) { return (a + b - 1) / b; };
  const int gblk = cdiv(N, 128);  // 8 waves x 16 rows per block

  auto attn = [&](const float* Aatt, int F, int Hh, int ldh) {
    int nh = N * Hh;
    k_node_dots<<<cdiv(nh, TB), TB, 0, stream>>>(hp, Aatt, dd1, dd2, N, F, Hh, ldh);
    k_fill_i32<<<cdiv(nh, TB), TB, 0, stream>>>(menc, ORD_NEG_INF, nh);
    (void)hipMemsetAsync(ssum, 0, (size_t)nh * 4, stream);
    (void)hipMemsetAsync(asum, 0, (size_t)nh * 4, stream);
    k_edge_max   <<<cdiv(E, TB), TB, 0, stream>>>(src, dst, dd1, dd2, menc, N, E, Hh);
    k_edge_expsum<<<cdiv(E, TB), TB, 0, stream>>>(src, dst, dd1, dd2, menc, ssum, N, E, Hh);
    k_edge_alpha <<<cdiv(E, TB), TB, 0, stream>>>(src, dst, dd1, dd2, menc, ssum, asum, N, E, Hh);
  };

  // ---- layer 0: x(N,128) -> hp(N,96) ----
  k_cvt_f32_bf16<<<cdiv(N * IN, TB), TB, 0, stream>>>(x, hb, N * IN);
  k_cvt_w_pad<<<cdiv(F0 * IN, TB), TB, 0, stream>>>(W0, wb, F0, F0, IN);
  k_gemm_tile<128, 6><<<gblk, 256, 0, stream>>>(hb, wb, hp, N, F0, F0);
  attn(A0, HID, H, F0);
  k_finalize_elu_bf16<<<cdiv(N * F0, TB), TB, 0, stream>>>(hp, asum, hb, N, HID, H);

  // ---- layer 1: h(N,96) -> hp(N,96) ----
  k_cvt_w_pad<<<cdiv(F0 * F0, TB), TB, 0, stream>>>(W1, wb, F0, F0, F0);
  k_gemm_tile<96, 6><<<gblk, 256, 0, stream>>>(hb, wb, hp, N, F0, F0);
  attn(A1, HID, H, F0);
  k_finalize_elu_bf16<<<cdiv(N * F0, TB), TB, 0, stream>>>(hp, asum, hb, N, HID, H);

  // ---- layer 2: h(N,96) -> logits(N,40) ----
  k_cvt_w_pad<<<cdiv(CLSP * F0, TB), TB, 0, stream>>>(W2, wb, CLS, CLSP, F0);
  k_gemm_tile<96, 3><<<gblk, 256, 0, stream>>>(hb, wb, hp, N, CLS, CLS);
  attn(A2, CLS, 1, CLS);
  k_finalize_out<<<cdiv(N * CLS, TB), TB, 0, stream>>>(hp, asum, (float*)d_out, N, CLS);
}
